// MSSE_Module_9251359556116
// MI455X (gfx1250) — matrix operand order repaired and checked by reading, not compiled, not run on hardware
//
#include <hip/hip_runtime.h>
#include <hip/hip_bf16.h>
#include <math.h>

typedef __attribute__((ext_vector_type(16))) _Float16 v16h;
typedef __attribute__((ext_vector_type(8)))  float    v8f;
typedef __attribute__((ext_vector_type(4)))  unsigned int u32x4;
typedef __attribute__((ext_vector_type(8)))  int      i32x8;
typedef __attribute__((ext_vector_type(4)))  int      i32x4;

#define TM 128      // output positions per block (8 waves x 16)
#define KSTEP 32    // K per WMMA step (f16 16x16x32)

#if defined(__has_builtin)
#if __has_builtin(__builtin_amdgcn_tensor_load_to_lds) && __has_builtin(__builtin_amdgcn_s_wait_tensorcnt)
#define HAVE_TDM 1
#endif
#endif
#ifndef HAVE_TDM
#define HAVE_TDM 0
#endif

__device__ __forceinline__ float mishf(float x) {
    float sp = (x > 20.f) ? x : log1pf(__expf(x));
    return x * tanhf(sp);
}

// ---------------------------------------------------------------------------
// Implicit-GEMM convolution, f16 WMMA with f32 accumulate.
//   M = N*H*H output positions, N-dim = output channels, K = C*KS*KS.
//   Weights pre-converted to f16, K padded to Kpad (multiple of 32).
//   B tile staged to LDS by the Tensor Data Mover (wave 0), A tile gathered.
//   flags: bit0 = BN(scale/shift), bit1 = mish, bit2 = accumulate into Y.
// ---------------------------------------------------------------------------
template<int NT, int KS>
__global__ __launch_bounds__(256)
void conv_wmma(const float* __restrict__ X, const _Float16* __restrict__ Wh,
               const float* __restrict__ bias,
               const float* __restrict__ scl, const float* __restrict__ shf,
               float* __restrict__ Y,
               int C, int H, int pad, int O, int Kpad, int outCtot, int chanOff,
               int flags)
{
    __shared__ _Float16 As[TM][KSTEP];   // [pos][k]    8 KB
    __shared__ _Float16 Bs[64][KSTEP];   // [chan][k]   4 KB (row = out-channel)

    const int tid  = threadIdx.x;
    const int wave = tid >> 5;
    const int lane = tid & 31;
    const unsigned HW = (unsigned)(H * H);
    constexpr int KK = KS * KS;
    const int Ktot = C * KK;
    const int posBase = blockIdx.x * TM;
    const int n0 = blockIdx.y * (NT * 16);

    // TM divides HW (16384 or 4096): whole block lives in one batch image.
    const unsigned bn   = (unsigned)posBase / HW;   // scalar
    const unsigned brem = (unsigned)posBase % HW;   // scalar
    const float* __restrict__ Xn = X + (size_t)bn * C * HW;

    // A-tile loader coords: 2 threads per row, 16 consecutive K each
    const int arow  = tid >> 1;
    const unsigned arem = brem + (unsigned)arow;
    const int aoh = (int)(arem / (unsigned)H);
    const int aow = (int)(arem % (unsigned)H);
    const int akoff = (tid & 1) * 16;

    v8f acc[NT];
#pragma unroll
    for (int t = 0; t < NT; ++t) {
        v8f z;
#pragma unroll
        for (int e = 0; e < 8; ++e) z[e] = 0.f;
        acc[t] = z;
    }

    for (int k0 = 0; k0 < Ktot; k0 += KSTEP) {
        // ---- stage B tile: TDM async DMA (wave 0) or manual packed copy ----
#if HAVE_TDM
        if (wave == 0) {
            const unsigned ldsb = (unsigned)(size_t)(&Bs[0][0]);
            const unsigned long long ga =
                (unsigned long long)(size_t)(Wh) + 2ull * ((size_t)n0 * Kpad + k0);
            u32x4 g0;
            g0[0] = 1u;                                   // count=1 (valid), user mode
            g0[1] = ldsb;                                 // lds_addr
            g0[2] = (unsigned)(ga & 0xffffffffu);         // global_addr[31:0]
            g0[3] = (unsigned)((ga >> 32) & 0x01ffffffu)  // global_addr[56:32]
                    | (2u << 30);                         // type=2 ("image")
            i32x8 g1;
            g1[0] = (int)(1u << 16);                      // data_size=1 (2 bytes)
            g1[1] = (int)(((unsigned)Kpad & 0xffffu) << 16);            // tensor_dim0[15:0]
            g1[2] = (int)((((unsigned)Kpad >> 16) & 0xffffu)            // tensor_dim0[31:16]
                    | (((unsigned)O & 0xffffu) << 16));                 // tensor_dim1[15:0]
            g1[3] = (int)((((unsigned)O >> 16) & 0xffffu)               // tensor_dim1[31:16]
                    | ((unsigned)KSTEP << 16));                         // tile_dim0=32
            g1[4] = (int)(NT * 16);                       // tile_dim1 rows; tile_dim2=0
            g1[5] = (int)Kpad;                            // tensor_dim0_stride[31:0]
            g1[6] = 0;                                    // stride0[47:32], stride1[15:0]
            g1[7] = 0;
            i32x4 zg4 = {0, 0, 0, 0};                     // D# groups 2/3 unused (2D tile)
            i32x8 zg8 = {0, 0, 0, 0, 0, 0, 0, 0};
            __builtin_amdgcn_tensor_load_to_lds(g0, g1, zg4, zg4, zg8, 0);
        }
#else
        {
            const unsigned* Wu = (const unsigned*)Wh;     // 2 halves per dword
#pragma unroll
            for (int i = 0; i < 4; ++i) {
                int idx = tid * 4 + i;                    // 1024 dwords total
                int row = idx >> 4;                       // 16 dwords per row
                int kkd = idx & 15;
                if (row < NT * 16)
                    ((unsigned*)Bs)[idx] =
                        Wu[((size_t)(n0 + row) * Kpad + k0) / 2 + kkd];
            }
        }
#endif
        // ---- stage A tile: gather (implicit GEMM), f32 -> f16 ----
#pragma unroll 4
        for (int i = 0; i < 16; ++i) {
            int k = k0 + akoff + i;
            float v = 0.f;
            if (k < Ktot) {
                int c  = k / KK;               // compile-time-constant divisors
                int r  = k - c * KK;
                int kh = r / KS;
                int kw = r - kh * KS;
                int ih = aoh + kh - pad;
                int iw = aow + kw - pad;
                if ((unsigned)ih < (unsigned)H && (unsigned)iw < (unsigned)H)
                    v = Xn[((size_t)c * H + ih) * H + iw];
            }
            As[arow][akoff + i] = (_Float16)v;
        }
#if HAVE_TDM
        if (wave == 0) __builtin_amdgcn_s_wait_tensorcnt(0);
#endif
        __syncthreads();

        // ---- A fragment (16x32 f16, documented wave32 layout) ----
        v16h afrag;
        {
            const int mrow = (wave << 4) + (lane & 15);
            const int kh8  = (lane >> 4) << 3;                 // 0 or 8
            const unsigned* ap = (const unsigned*)(&As[mrow][0]);
            unsigned* au = (unsigned*)&afrag;
#pragma unroll
            for (int j = 0; j < 4; ++j) au[j]     = ap[(kh8 >> 1) + j];        // K=kh8..kh8+7
#pragma unroll
            for (int j = 0; j < 4; ++j) au[4 + j] = ap[((kh8 + 16) >> 1) + j]; // K=kh8+16..
        }
        const int bcol0 = lane & 15;
        const int kb    = (lane >> 4) << 3;                    // 0 or 8
#pragma unroll
        for (int nt = 0; nt < NT; ++nt) {
            v16h bfrag;
            const unsigned* bp = (const unsigned*)(&Bs[nt * 16 + bcol0][0]);
            unsigned* bu = (unsigned*)&bfrag;
#pragma unroll
            for (int j = 0; j < 4; ++j) bu[j] = bp[(kb >> 1) + j];             // K=kb..kb+7
#pragma unroll
            for (int j = 0; j < 4; ++j) bu[4 + j] = bp[((kb + 16) >> 1) + j];  // K=kb+16..kb+23
            acc[nt] = __builtin_amdgcn_wmma_f32_16x16x32_f16(
                false, afrag, false, bfrag, (short)0, acc[nt], false, false);
        }
        __syncthreads();
    }

    // ---- epilogue: bias (+BN)(+mish), store/accumulate ----
    const int mbase = (wave << 4) + ((lane >> 4) << 3);
    float* __restrict__ Yn = Y + ((size_t)bn * outCtot + chanOff) * HW;
#pragma unroll
    for (int nt = 0; nt < NT; ++nt) {
        const int ch = n0 + nt * 16 + (lane & 15);
        const float bz = bias[ch];
        float sc = 1.f, sh = 0.f;
        if (flags & 1) { sc = scl[ch]; sh = shf[ch]; }
#pragma unroll
        for (int i = 0; i < 8; ++i) {
            unsigned rem = brem + (unsigned)(mbase + i);
            float v = acc[nt][i] + bz;
            if (flags & 1) v = v * sc + sh;
            if (flags & 2) v = mishf(v);
            size_t oidx = (size_t)ch * HW + rem;
            if (flags & 4) Yn[oidx] += v;
            else           Yn[oidx] = v;
        }
    }
}

// ---------------------------------------------------------------------------
// Weight prep: fp32 OIHW [O][Ktot] -> f16 [O][Kpad], zero-padded K tail
// ---------------------------------------------------------------------------
__global__ void wprep(const float* __restrict__ w, _Float16* __restrict__ o,
                      int Ktot, int Kpad, int total) {
    int idx = blockIdx.x * blockDim.x + threadIdx.x;
    if (idx >= total) return;
    int row = idx / Kpad;
    int k   = idx - row * Kpad;
    o[idx] = (k < Ktot) ? (_Float16)w[(size_t)row * Ktot + k] : (_Float16)0.f;
}

// BN eval-mode scale/shift precompute
__global__ void bn_prep(const float* g, const float* b, const float* m,
                        const float* v, float* scl, float* shf, int n) {
    int i = blockIdx.x * blockDim.x + threadIdx.x;
    if (i < n) {
        float s = g[i] * rsqrtf(v[i] + 1e-5f);
        scl[i] = s;
        shf[i] = b[i] - m[i] * s;
    }
}

// Global average pool over H*W; one block per (n,c)
__global__ __launch_bounds__(256)
void avgpool_hw(const float* __restrict__ in, float* __restrict__ out, int HW) {
    __shared__ float red[256];
    const float* p = in + (size_t)blockIdx.x * HW;
    float s = 0.f;
    for (int i = threadIdx.x; i < HW; i += 256) s += p[i];
    red[threadIdx.x] = s;
    __syncthreads();
    for (int off = 128; off > 0; off >>= 1) {
        if ((int)threadIdx.x < off) red[threadIdx.x] += red[threadIdx.x + off];
        __syncthreads();
    }
    if (threadIdx.x == 0) out[blockIdx.x] = red[0] / (float)HW;
}

// out[n][o] = sigmoid(dot(pooled[n], fw[o]) + fb[o])
__global__ void fc_sigmoid(const float* __restrict__ pooled,
                           const float* __restrict__ fw, const float* __restrict__ fb,
                           float* __restrict__ out, int NB, int C) {
    int id = blockIdx.x * blockDim.x + threadIdx.x;
    if (id >= NB * C) return;
    int n = id / C, o = id % C;
    float s = fb[o];
    for (int c = 0; c < C; ++c) s += pooled[n * C + c] * fw[o * C + c];
    out[id] = 1.f / (1.f + __expf(-s));
}

// res_io = att[n,c] * conn + res_io   (in place)
__global__ void scale_residual(const float* __restrict__ att,
                               const float* __restrict__ conn,
                               float* __restrict__ res_io, int HW, size_t total) {
    size_t idx = (size_t)blockIdx.x * blockDim.x + threadIdx.x;
    if (idx >= total) return;
    size_t nc = idx / (unsigned)HW;
    res_io[idx] = att[nc] * conn[idx] + res_io[idx];
}

// Haar DWT (zero mode, even dims): in [NB][C][2Ho][2Ho] ->
//   xlow = cA [NB][C][Ho][Ho];  high = [NB][3C][Ho][Ho] as [cH | cV | cD]
__global__ void haar_dwt(const float* __restrict__ in, float* __restrict__ xlow,
                         float* __restrict__ high, int C, int Ho, size_t total) {
    size_t idx = (size_t)blockIdx.x * blockDim.x + threadIdx.x;
    if (idx >= total) return;
    unsigned ow = (unsigned)(idx % (unsigned)Ho);
    size_t t    = idx / (unsigned)Ho;
    unsigned oh = (unsigned)(t % (unsigned)Ho);
    size_t nc   = t / (unsigned)Ho;
    unsigned n  = (unsigned)(nc / (unsigned)C);
    unsigned c  = (unsigned)(nc % (unsigned)C);
    const int Hi = 2 * Ho;
    size_t ibase = ((nc * Hi) + 2 * oh) * Hi + 2 * ow;
    float x00 = in[ibase],       x01 = in[ibase + 1];
    float x10 = in[ibase + Hi],  x11 = in[ibase + Hi + 1];
    float cA = 0.5f * (x00 + x01 + x10 + x11);
    float cH = 0.5f * (x00 + x01 - x10 - x11);
    float cV = 0.5f * (x00 - x01 + x10 - x11);
    float cD = 0.5f * (x00 - x01 - x10 + x11);
    xlow[idx] = cA;
    size_t plane = (size_t)Ho * Ho;
    size_t hb = (((size_t)n * (3 * C) + c) * plane) + (size_t)oh * Ho + ow;
    high[hb]                         = cH;
    high[hb + (size_t)C * plane]     = cV;
    high[hb + (size_t)2 * C * plane] = cD;
}

// ---------------------------------------------------------------------------
// Input ordering (setup_inputs dict insertion order, nested dicts flattened)
// ---------------------------------------------------------------------------
enum {
    IN_X = 0,
    IN_RES_W, IN_RES_B,
    IN_C1_W, IN_C1_B, IN_BN1_G, IN_BN1_B, IN_BN1_M, IN_BN1_V,
    IN_C2_W, IN_C2_B, IN_BN2_G, IN_BN2_B, IN_BN2_M, IN_BN2_V,
    IN_ATT_W, IN_ATT_B, IN_ATTBN_G, IN_ATTBN_B, IN_ATTBN_M, IN_ATTBN_V,
    IN_FC_W, IN_FC_B,
    IN_P1_W, IN_P1_B, IN_P2_W, IN_P2_B, IN_P3_W, IN_P3_B,
    IN_SKIP_W, IN_SKIP_B, IN_SPLIT_W, IN_SPLIT_B,
    IN_HI_W, IN_HI_B, IN_HIBN_G, IN_HIBN_B, IN_HIBN_M, IN_HIBN_V
};

extern "C" void kernel_launch(void* const* d_in, const int* in_sizes, int n_in,
                              void* d_out, int out_size, void* d_ws, size_t ws_size,
                              hipStream_t stream) {
    (void)in_sizes; (void)n_in; (void)out_size; (void)ws_size;

    const float* X = (const float*)d_in[IN_X];
    float* out = (float*)d_out;
    float* ws  = (float*)d_ws;

    const int NB = 16, CIN = 32, H = 128, HW = H * H, CO = 64, Ho = 64;
    const size_t NHW  = (size_t)NB * HW;        // 262144 positions
    const size_t NHWo = (size_t)NB * Ho * Ho;   // 65536 positions

    // workspace layout (floats)
    float* res    = ws;                          // [16][64][128][128]   16777216
    float* abuf   = ws + 16777216ull;            // att-out / split-out  16777216
    float* c1buf  = ws + 33554432ull;            // [16][16][128][128]    4194304
    float* concat = ws + 37748736ull;            // [16][192][128][128]  50331648 (reused as DWT-high)
    float* pooled = ws + 88080384ull;            // [16][64]
    float* attv   = ws + 88081408ull;            // [16][64]
    float* bnbuf  = ws + 88082432ull;            // BN scale/shift (512)
    float *bn1_s = bnbuf,        *bn1_h = bnbuf + 16;
    float *bn2_s = bnbuf + 32,   *bn2_h = bnbuf + 96;
    float *abn_s = bnbuf + 160,  *abn_h = bnbuf + 224;
    float *hbn_s = bnbuf + 288,  *hbn_h = bnbuf + 352;

    // f16 weight arena (in float units, 2 halves per float)
    _Float16* warena = (_Float16*)(ws + 88082944ull);
    struct WInfo { int in_idx, O, Ktot, Kpad; };
    const WInfo wi[10] = {
        { IN_RES_W,   64,   32,   32 },   // 0 res    1x1
        { IN_C1_W,    16,  288,  288 },   // 1 c1     3x3
        { IN_C2_W,    64,  144,  160 },   // 2 c2     3x3 (K padded)
        { IN_ATT_W,   64,   64,   64 },   // 3 att    1x1
        { IN_P1_W,    64,  576,  576 },   // 4 p1     3x3
        { IN_P2_W,    64, 1600, 1600 },   // 5 p2     5x5
        { IN_P3_W,    64, 3136, 3136 },   // 6 p3     7x7
        { IN_SKIP_W, 192,   32,   32 },   // 7 skip   1x1
        { IN_SPLIT_W, 64,  192,  192 },   // 8 split  1x1
        { IN_HI_W,    64,  192,  192 },   // 9 hi     1x1
    };
    _Float16* whp[10];
    {
        size_t off = 0;
        for (int i = 0; i < 10; ++i) {
            whp[i] = warena + off;
            int total = wi[i].O * wi[i].Kpad;
            wprep<<<(total + 255) / 256, 256, 0, stream>>>(
                (const float*)d_in[wi[i].in_idx], whp[i], wi[i].Ktot, wi[i].Kpad, total);
            off += (size_t)total;
        }
    }

    // output regions: (x_low, x_high, conn_x) concatenated
    float* xlow  = out;                          // 16*64*64*64   = 4194304
    float* xhigh = out + 4194304ull;             // 16*64*64*64   = 4194304
    float* conn  = out + 8388608ull;             // 16*64*128*128 = 16777216

    dim3 blk(256);

    bn_prep<<<1, 64, 0, stream>>>((const float*)d_in[IN_BN1_G], (const float*)d_in[IN_BN1_B],
                                  (const float*)d_in[IN_BN1_M], (const float*)d_in[IN_BN1_V],
                                  bn1_s, bn1_h, 16);
    bn_prep<<<1, 64, 0, stream>>>((const float*)d_in[IN_BN2_G], (const float*)d_in[IN_BN2_B],
                                  (const float*)d_in[IN_BN2_M], (const float*)d_in[IN_BN2_V],
                                  bn2_s, bn2_h, 64);
    bn_prep<<<1, 64, 0, stream>>>((const float*)d_in[IN_ATTBN_G], (const float*)d_in[IN_ATTBN_B],
                                  (const float*)d_in[IN_ATTBN_M], (const float*)d_in[IN_ATTBN_V],
                                  abn_s, abn_h, 64);
    bn_prep<<<1, 64, 0, stream>>>((const float*)d_in[IN_HIBN_G], (const float*)d_in[IN_HIBN_B],
                                  (const float*)d_in[IN_HIBN_M], (const float*)d_in[IN_HIBN_V],
                                  hbn_s, hbn_h, 64);

    const int gx  = (int)(NHW / TM);    // 2048
    const int gxo = (int)(NHWo / TM);   // 512

    // residual = conv1x1(x, 32->64) + bias
    conv_wmma<4, 1><<<dim3(gx, 1), blk, 0, stream>>>(X, whp[0],
        (const float*)d_in[IN_RES_B], nullptr, nullptr, res,
        CIN, H, 0, 64, wi[0].Kpad, CO, 0, 0);

    // c1 = mish(BN(conv3x3(x, 32->16)))
    conv_wmma<1, 3><<<dim3(gx, 1), blk, 0, stream>>>(X, whp[1],
        (const float*)d_in[IN_C1_B], bn1_s, bn1_h, c1buf,
        CIN, H, 1, 16, wi[1].Kpad, 16, 0, 1 | 2);

    // conn_x = mish(BN(conv3x3(c1, 16->64)))  -> written straight into d_out
    conv_wmma<4, 3><<<dim3(gx, 1), blk, 0, stream>>>(c1buf, whp[2],
        (const float*)d_in[IN_C2_B], bn2_s, bn2_h, conn,
        16, H, 1, 64, wi[2].Kpad, CO, 0, 1 | 2);

    // attention: a = mish(BN(conv1x1(conn, 64->64)))
    conv_wmma<4, 1><<<dim3(gx, 1), blk, 0, stream>>>(conn, whp[3],
        (const float*)d_in[IN_ATT_B], abn_s, abn_h, abuf,
        CO, H, 0, 64, wi[3].Kpad, CO, 0, 1 | 2);
    avgpool_hw<<<NB * CO, 256, 0, stream>>>(abuf, pooled, HW);
    fc_sigmoid<<<(NB * CO + 255) / 256, 256, 0, stream>>>(pooled,
        (const float*)d_in[IN_FC_W], (const float*)d_in[IN_FC_B], attv, NB, CO);

    // res <- att * conn + res   (in place)
    scale_residual<<<65536, 256, 0, stream>>>(attv, conn, res, HW, (size_t)NB * CO * HW);

    // multi-scale convs into concat[192]
    conv_wmma<4, 3><<<dim3(gx, 1), blk, 0, stream>>>(res, whp[4],
        (const float*)d_in[IN_P1_B], nullptr, nullptr, concat,
        CO, H, 1, 64, wi[4].Kpad, 192, 0, 2);
    conv_wmma<4, 5><<<dim3(gx, 1), blk, 0, stream>>>(res, whp[5],
        (const float*)d_in[IN_P2_B], nullptr, nullptr, concat,
        CO, H, 2, 64, wi[5].Kpad, 192, 64, 2);
    conv_wmma<4, 7><<<dim3(gx, 1), blk, 0, stream>>>(res, whp[6],
        (const float*)d_in[IN_P3_B], nullptr, nullptr, concat,
        CO, H, 3, 64, wi[6].Kpad, 192, 128, 2);

    // concat += conv1x1(x, 32->192)  (fused accumulate epilogue)
    conv_wmma<4, 1><<<dim3(gx, 3), blk, 0, stream>>>(X, whp[7],
        (const float*)d_in[IN_SKIP_B], nullptr, nullptr, concat,
        CIN, H, 0, 192, wi[7].Kpad, 192, 0, 4);

    // split: conv1x1(concat, 192->64) -> abuf (reuse)
    conv_wmma<4, 1><<<dim3(gx, 1), blk, 0, stream>>>(concat, whp[8],
        (const float*)d_in[IN_SPLIT_B], nullptr, nullptr, abuf,
        192, H, 0, 64, wi[8].Kpad, CO, 0, 0);

    // Haar DWT: abuf -> x_low (d_out) + high[192ch] (reuse concat)
    haar_dwt<<<(unsigned)((NHWo * CO + 255) / 256), 256, 0, stream>>>(
        abuf, xlow, concat, CO, Ho, (size_t)NB * CO * Ho * Ho);

    // x_high = mish(BN(conv1x1(high, 192->64)))  at 64x64
    conv_wmma<4, 1><<<dim3(gxo, 1), blk, 0, stream>>>(concat, whp[9],
        (const float*)d_in[IN_HI_B], hbn_s, hbn_h, xhigh,
        192, Ho, 0, 64, wi[9].Kpad, CO, 0, 1 | 2);
}
